// MLPDecoder_68289980006959
// MI455X (gfx1250) — matrix operand order repaired and checked by reading, not compiled, not run on hardware
//
#include <hip/hip_runtime.h>
#include <hip/hip_bf16.h>

typedef __attribute__((ext_vector_type(16))) _Float16 v16h;
typedef __attribute__((ext_vector_type(8)))  _Float16 v8h;
typedef __attribute__((ext_vector_type(2)))  _Float16 v2h;
typedef __attribute__((ext_vector_type(8)))  float    v8f;

#define WMMA_F32_F16(a, b, c) \
  __builtin_amdgcn_wmma_f32_16x16x32_f16(false, (a), false, (b), (short)0, (c), false, false)

__device__ __forceinline__ v16h cat8(v8h lo, v8h hi) {
  return __builtin_shufflevector(lo, hi, 0, 1, 2, 3, 4, 5, 6, 7, 8, 9, 10, 11, 12, 13, 14, 15);
}

// ---- fragment loaders ------------------------------------------------------
// A fragment (M x K, row-major, stride ld halves). ISA: lane&15 = row M,
// K split kb = (lane>>4)*8: V0-3 hold K=kb..kb+7, V4-7 hold K=16+kb..16+kb+7.
__device__ __forceinline__ v16h fragA_k32(const _Float16* base, int ld, int lane) {
  const int row = lane & 15;
  const int kb  = (lane >> 4) << 3;
  v8h lo = *(const v8h*)(base + row * ld + kb);
  v8h hi = *(const v8h*)(base + row * ld + 16 + kb);
  return cat8(lo, hi);
}
// A fragment, K=16 real data (upper K zero-padded).
__device__ __forceinline__ v16h fragA_k16(const _Float16* base, int ld, int lane) {
  const int row = lane & 15;
  const int kb  = (lane >> 4) << 3;
  v8h lo = *(const v8h*)(base + row * ld + kb);
  v8h z  = {};
  return cat8(lo, z);
}
// B fragment from B^T (N x K, row-major, stride ld halves). ISA: lane&15 = col N,
// K split kb = (lane>>4)*8: V0-3 hold K=kb..kb+7, V4-7 hold K=16+kb..16+kb+7.
__device__ __forceinline__ v16h fragB_k32(const _Float16* base, int ld, int lane) {
  const int row = lane & 15;
  const int kb  = (lane >> 4) << 3;
  v8h lo = *(const v8h*)(base + row * ld + kb);
  v8h hi = *(const v8h*)(base + row * ld + 16 + kb);
  return cat8(lo, hi);
}

// bias + ELU + convert one 16x16 f32 C tile row-slice to f16
__device__ __forceinline__ v8h elu_h8(v8f c, v8f bias) {
  v8h r;
#pragma unroll
  for (int i = 0; i < 8; ++i) {
    const float v = c[i] + bias[i];
    r[i] = (_Float16)(v > 0.f ? v : (__expf(v) - 1.f));
  }
  return r;
}

// ---- prep: initial state + f16 transposed weight packs ---------------------
// W1T: (3,128,32) K zero-padded 16->32.  W2T: (3,128,128).
// Wo1T: (128,160) K padded 136->160.     Wo2T: (16,128) N padded 8->16.
__global__ void prep_kernel(const float* __restrict__ x,
                            const float* __restrict__ W1, const float* __restrict__ W2,
                            const float* __restrict__ Wo1, const float* __restrict__ Wo2,
                            float* __restrict__ last0,
                            _Float16* __restrict__ W1T, _Float16* __restrict__ W2T,
                            _Float16* __restrict__ Wo1T, _Float16* __restrict__ Wo2T) {
  const int gs  = blockDim.x * gridDim.x;
  const int id0 = blockIdx.x * blockDim.x + threadIdx.x;
  for (int idx = id0; idx < 8 * 10 * 64 * 8; idx += gs) {  // last0[bt][n][d] = x[b][n][t*10][d]
    const int d = idx & 7, n = (idx >> 3) & 63, bt = idx >> 9;
    const int t = bt % 10, b = bt / 10;
    last0[idx] = x[(((size_t)b * 64 + n) * 100 + t * 10) * 8 + d];
  }
  for (int idx = id0; idx < 3 * 128 * 32; idx += gs) {  // W1T[k][n][g] = W1[1+k][g][n]
    const int g = idx & 31, n = (idx >> 5) & 127, k = idx >> 12;
    W1T[idx] = (_Float16)((g < 16) ? W1[((size_t)(1 + k) * 16 + g) * 128 + n] : 0.f);
  }
  for (int idx = id0; idx < 3 * 128 * 128; idx += gs) {  // W2T[k][n][g] = W2[1+k][g][n]
    const int g = idx & 127, n = (idx >> 7) & 127, k = idx >> 14;
    W2T[idx] = (_Float16)W2[((size_t)(1 + k) * 128 + g) * 128 + n];
  }
  for (int idx = id0; idx < 128 * 160; idx += gs) {  // Wo1T[n][g]
    const int g = idx % 160, n = idx / 160;
    Wo1T[idx] = (_Float16)((g < 136) ? Wo1[(size_t)g * 128 + n] : 0.f);
  }
  for (int idx = id0; idx < 16 * 128; idx += gs) {  // Wo2T[n][g]
    const int g = idx & 127, n = idx >> 7;
    Wo2T[idx] = (_Float16)((n < 8) ? Wo2[(size_t)g * 8 + n] : 0.f);
  }
}

// ---- per-receiver edge MLP + weighted aggregation --------------------------
// grid = 80*64 blocks; block = (bt, receiver r); 128 threads = 4 waves.
// Each wave owns 16 of the 64 (63 real) edge rows. First GEMM is computed
// TRANSPOSED so the second GEMM's A fragments are built entirely in registers.
__global__ __launch_bounds__(128) void edge_aggr_kernel(
    const float* __restrict__ last, const float* __restrict__ edges,
    const _Float16* __restrict__ W1T, const float* __restrict__ b1,
    const _Float16* __restrict__ W2T, const float* __restrict__ b2,
    float* __restrict__ aggr) {
  const int blk = blockIdx.x;
  const int r   = blk & 63;
  const int bt  = blk >> 6;  // 0..79
  const int b   = bt / 10;
  const float* lastbt = last + (size_t)bt * (64 * 8);

  __shared__ __align__(16) _Float16 inpL[64 * 32];  // [edge][2D=16 | 16 zero-pad]
  __shared__ float aggrL[128];

  const int tid = threadIdx.x;
  for (int idx = tid; idx < 64 * 8; idx += 128) {
    const int j = idx >> 3, d = idx & 7;
    float vs = 0.f, vr = 0.f;
    if (j < 63) {
      const int s = j + (j >= r);  // send index skips diagonal
      vs = lastbt[s * 8 + d];
      vr = lastbt[r * 8 + d];
    }
    inpL[j * 32 + d]     = (_Float16)vs;
    inpL[j * 32 + 8 + d] = (_Float16)vr;
  }
  for (int idx = tid; idx < 64 * 8; idx += 128) {  // zero K pad (pairs)
    const int j = idx >> 3, d = idx & 7;
    *(v2h*)(inpL + j * 32 + 16 + 2 * d) = (v2h){};
  }
  aggrL[tid] = 0.f;
  __syncthreads();

  const int wv = tid >> 5, lane = tid & 31;
  const int colq = lane & 15;
  const int rb   = (lane >> 4) << 3;

  // B operand of transposed first GEMM: this wave's 16 edges x K (zero-padded)
  const v16h bI = fragB_k32(inpL + wv * (16 * 32), 32, lane);

  // per-lane edge weights for rows erow0..erow0+7 (k = 1..3 of edges)
  const float* eb = edges + ((size_t)b * 4032 + (size_t)r * 63) * 4;
  const int erow0 = wv * 16 + rb;
  float wreg[3][8];
#pragma unroll
  for (int i = 0; i < 8; ++i) {
    const int j = erow0 + i;
#pragma unroll
    for (int k = 0; k < 3; ++k) wreg[k][i] = (j < 63) ? eb[j * 4 + 1 + k] : 0.f;
  }

  float acc_s[8];
#pragma unroll
  for (int i = 0; i < 8; ++i) acc_s[i] = 0.f;

  for (int k = 0; k < 3; ++k) {
    // hT tiles: rows g = 16*nt+rb+i, cols e = colq   (A = W1T, B = inp)
    v8h hv[8];
#pragma unroll
    for (int nt = 0; nt < 8; ++nt) {
      v8f c = {};
      const v16h aW = fragA_k16(W1T + (size_t)(k * 128 + nt * 16) * 32, 32, lane);
      c = WMMA_F32_F16(aW, bI, c);
      const v8f bias = *(const v8f*)(b1 + (1 + k) * 128 + nt * 16 + rb);
      hv[nt] = elu_h8(c, bias);  // lane now holds h[g=16nt+rb+i][e=colq]
    }
    // msg = h @ W2[k] + b2[k]; A fragments assembled from hv in registers
#pragma unroll
    for (int nt = 0; nt < 8; ++nt) {
      const float bias2 = b2[(1 + k) * 128 + nt * 16 + colq];
      v8f c;
#pragma unroll
      for (int i = 0; i < 8; ++i) c[i] = bias2;
#pragma unroll
      for (int kk = 0; kk < 4; ++kk) {
        const v16h aH = cat8(hv[2 * kk], hv[2 * kk + 1]);
        const v16h bW = fragB_k32(W2T + (size_t)(k * 128 + nt * 16) * 128 + 32 * kk, 128, lane);
        c = WMMA_F32_F16(aH, bW, c);
      }
      float s = 0.f;  // weighted reduction over this lane's 8 edge rows
#pragma unroll
      for (int i = 0; i < 8; ++i) s += wreg[k][i] * c[i];
      acc_s[nt] += s;
    }
  }
#pragma unroll
  for (int nt = 0; nt < 8; ++nt) atomicAdd(&aggrL[nt * 16 + colq], acc_s[nt]);
  __syncthreads();
  aggr[(size_t)blk * 128 + tid] = aggrL[tid];
}

// ---- per-node output MLP + residual + prediction write ---------------------
// grid = 80 blocks; 128 threads = 4 waves (16 nodes each). Hidden layer is
// computed transposed -> second GEMM A fragments stay in registers.
__global__ __launch_bounds__(128) void node_update_kernel(
    const float* __restrict__ aggr, const float* __restrict__ last,
    const _Float16* __restrict__ Wo1T, const float* __restrict__ bo1,
    const _Float16* __restrict__ Wo2T, const float* __restrict__ bo2,
    float* __restrict__ last_next, float* __restrict__ out, int step) {
  const int bt  = blockIdx.x;  // 0..79
  const int tid = threadIdx.x, wv = tid >> 5, lane = tid & 31;
  __shared__ __align__(16) _Float16 aiL[64 * 160];  // [aggr(128) | last(8) | pad]
  const float* agg = aggr + (size_t)bt * (64 * 128);
  const float* lst = last + (size_t)bt * (64 * 8);

  for (int idx = tid; idx < 64 * 80; idx += 128) {  // packed 2-half stores
    const int n = idx / 80, cp = (idx - n * 80) * 2;
    v2h p = {};
    if (cp < 128) {
      p[0] = (_Float16)agg[n * 128 + cp];
      p[1] = (_Float16)agg[n * 128 + cp + 1];
    } else if (cp < 136) {
      p[0] = (_Float16)lst[n * 8 + cp - 128];
      p[1] = (_Float16)lst[n * 8 + cp - 127];
    }
    *(v2h*)(aiL + n * 160 + cp) = p;
  }
  __syncthreads();

  const _Float16* aiW = aiL + wv * (16 * 160);
  const int colq = lane & 15, rb = (lane >> 4) << 3;

  // B fragments of transposed hidden GEMM: this wave's 16 nodes, K=160
  v16h bAi[5];
#pragma unroll
  for (int kk = 0; kk < 5; ++kk) bAi[kk] = fragB_k32(aiW + 32 * kk, 160, lane);

  // hidT tiles: rows g = 16*nt+rb+i, cols node = colq
  v8h hv[8];
#pragma unroll
  for (int nt = 0; nt < 8; ++nt) {
    v8f c = {};
#pragma unroll
    for (int kk = 0; kk < 5; ++kk) {
      const v16h aW = fragA_k32(Wo1T + (size_t)(nt * 16) * 160 + 32 * kk, 160, lane);
      c = WMMA_F32_F16(aW, bAi[kk], c);
    }
    const v8f bias = *(const v8f*)(bo1 + nt * 16 + rb);
    hv[nt] = elu_h8(c, bias);
  }

  // out = hid @ Wo2 : D rows = node (rb+i), cols = d (colq, <8 valid)
  v8f c = {};
#pragma unroll
  for (int kk = 0; kk < 4; ++kk) {
    const v16h aH = cat8(hv[2 * kk], hv[2 * kk + 1]);
    const v16h bW = fragB_k32(Wo2T + 32 * kk, 128, lane);
    c = WMMA_F32_F16(aH, bW, c);
  }
  const int bb = bt / 10, t = bt - bb * 10;
  const int time = t * 10 + step;
  if (colq < 8) {
#pragma unroll
    for (int i = 0; i < 8; ++i) {
      const int node = wv * 16 + rb + i;
      const float v = c[i] + bo2[colq] + lst[node * 8 + colq];  // residual
      last_next[(size_t)bt * (64 * 8) + node * 8 + colq] = v;
      if (time < 99)  // output drops final timestep
        out[(((size_t)bb * 64 + node) * 99 + time) * 8 + colq] = v;
    }
  }
}

// ---- launch ----------------------------------------------------------------
extern "C" void kernel_launch(void* const* d_in, const int* in_sizes, int n_in,
                              void* d_out, int out_size, void* d_ws, size_t ws_size,
                              hipStream_t stream) {
  const float* x     = (const float*)d_in[0];
  const float* edges = (const float*)d_in[1];
  const float* W1    = (const float*)d_in[4];
  const float* b1    = (const float*)d_in[5];
  const float* W2    = (const float*)d_in[6];
  const float* b2    = (const float*)d_in[7];
  const float* Wo1   = (const float*)d_in[8];
  const float* bo1   = (const float*)d_in[9];
  const float* Wo2   = (const float*)d_in[10];
  const float* bo2   = (const float*)d_in[11];
  float* out = (float*)d_out;

  char* ws = (char*)d_ws;
  size_t off = 0;
  auto wsalloc = [&](size_t bytes) -> char* {
    char* p = ws + off;
    off = (off + bytes + 255) & ~(size_t)255;
    return p;
  };
  float*    lastA = (float*)wsalloc(8 * 10 * 64 * 8 * sizeof(float));
  float*    lastB = (float*)wsalloc(8 * 10 * 64 * 8 * sizeof(float));
  float*    aggr  = (float*)wsalloc(80ull * 64 * 128 * sizeof(float));
  _Float16* W1T   = (_Float16*)wsalloc(3 * 128 * 32 * sizeof(_Float16));
  _Float16* W2T   = (_Float16*)wsalloc(3 * 128 * 128 * sizeof(_Float16));
  _Float16* Wo1T  = (_Float16*)wsalloc(128 * 160 * sizeof(_Float16));
  _Float16* Wo2T  = (_Float16*)wsalloc(16 * 128 * sizeof(_Float16));

  prep_kernel<<<120, 256, 0, stream>>>(x, W1, W2, Wo1, Wo2, lastA, W1T, W2T, Wo1T, Wo2T);

  float* cur = lastA;
  float* nxt = lastB;
  for (int step = 0; step < 10; ++step) {
    edge_aggr_kernel<<<80 * 64, 128, 0, stream>>>(cur, edges, W1T, b1, W2T, b2, aggr);
    node_update_kernel<<<80, 128, 0, stream>>>(aggr, cur, Wo1T, bo1, Wo2T, bo2, nxt, out, step);
    float* tmp = cur; cur = nxt; nxt = tmp;
  }
}
